// Attention_35691178230096
// MI455X (gfx1250) — compile-verified, matrix operand order checked by reading, not run on hardware
//
#include <hip/hip_runtime.h>
#include <hip/hip_bf16.h>

// Problem constants (reference: B=4, N=2048, DIM=512, H=8, DH=64)
#define B_    4
#define N_    2048
#define DIM_  512
#define H_    8
#define DH_   64
#define BN_   (B_ * N_)          // 8192 tokens
#define QKVN_ (3 * DIM_)         // 1536

typedef __bf16          v16bf __attribute__((ext_vector_type(16)));
typedef float           v8f   __attribute__((ext_vector_type(8)));
typedef unsigned short  u16x8 __attribute__((ext_vector_type(8)));

union Op16 { v16bf v; u16x8 h[2]; unsigned short s[16]; };

__device__ __forceinline__ unsigned short f2bf(float f) {
  unsigned int u = __float_as_uint(f);
  return (unsigned short)((u + 0x7FFFu + ((u >> 16) & 1u)) >> 16);  // RNE
}
__device__ __forceinline__ float bf2f(unsigned short s) {
  return __uint_as_float(((unsigned int)s) << 16);
}

// Load a bf16 WMMA operand whose K-run is contiguous in memory.
// base = &mat[row_or_col][k0]; lane layout: lanes 0-15 take K 0-7 & 16-23,
// lanes 16-31 take K 8-15 & 24-31 (ISA 16-bit A/B layout, 16x16x32).
__device__ __forceinline__ v16bf load_frag(const unsigned short* base, int hsel) {
  Op16 o;
  const unsigned short* p = base + hsel * 8;
  o.h[0] = *(const u16x8*)(p);
  o.h[1] = *(const u16x8*)(p + 16);
  return o.v;
}

// ---------------------------------------------------------------------------
// Kernel 0: convert x -> bf16 (row major), transpose+convert w_qkv, w_out so
// their WMMA B-operands are K-contiguous ([n][k] layout).
// ---------------------------------------------------------------------------
__global__ __launch_bounds__(256) void prep_kernel(
    const float* __restrict__ x, const float* __restrict__ wqkv,
    const float* __restrict__ wout,
    unsigned short* __restrict__ xbf, unsigned short* __restrict__ wqkvT,
    unsigned short* __restrict__ woutT) {
  const int tid    = blockIdx.x * blockDim.x + threadIdx.x;
  const int stride = gridDim.x * blockDim.x;
  for (int i = tid; i < BN_ * DIM_; i += stride) xbf[i] = f2bf(x[i]);
  for (int i = tid; i < DIM_ * QKVN_; i += stride) {
    const int k = i / QKVN_, n = i % QKVN_;
    wqkvT[n * DIM_ + k] = f2bf(wqkv[i]);
  }
  for (int i = tid; i < DIM_ * DIM_; i += stride) {
    const int k = i >> 9, n = i & (DIM_ - 1);
    woutT[n * DIM_ + k] = f2bf(wout[i]);
  }
}

// ---------------------------------------------------------------------------
// Kernel 1: QKV projection GEMM [8192x512]x[512x1536]+bias, bf16 WMMA.
// Each wave owns a 16x64 output tile (4 accumulators). Scatters into
// Q[b,h,n,d] (pre-scaled by 1/sqrt(DH)), K[b,h,n,d], V^T[b,h,d,n] as bf16.
// ---------------------------------------------------------------------------
__global__ __launch_bounds__(256) void qkv_gemm(
    const unsigned short* __restrict__ xbf,
    const unsigned short* __restrict__ wT,
    const float* __restrict__ bias,
    unsigned short* __restrict__ qws, unsigned short* __restrict__ kws,
    unsigned short* __restrict__ vtws) {
  const int wave = threadIdx.x >> 5;
  const int lane = threadIdx.x & 31;
  const int hsel = lane >> 4;
  const int row  = lane & 15;
  const int tid  = blockIdx.x * 8 + wave;
  const int m0   = (tid / 24) * 16;   // 512 M-tiles
  const int n0   = (tid % 24) * 64;   // 24 N-tiles

  v8f acc[4] = {};
  const unsigned short* arow = xbf + (size_t)(m0 + row) * DIM_;
  for (int k0 = 0; k0 < DIM_; k0 += 32) {
    v16bf a = load_frag(arow + k0, hsel);
#pragma unroll
    for (int sub = 0; sub < 4; ++sub) {
      v16bf b = load_frag(wT + (size_t)(n0 + sub * 16 + row) * DIM_ + k0, hsel);
      acc[sub] = __builtin_amdgcn_wmma_f32_16x16x32_bf16(
          false, a, false, b, (short)0, acc[sub], false, false);
    }
  }
#pragma unroll
  for (int sub = 0; sub < 4; ++sub) {
    const int c  = n0 + sub * 16 + row;
    const float bv = bias[c];
    const int t  = c / DIM_;      // 0=q 1=k 2=v
    const int hc = c % DIM_;
    const int hh = hc >> 6, d = hc & 63;
#pragma unroll
    for (int r = 0; r < 8; ++r) {
      const int tok = m0 + r + hsel * 8;      // D layout: M = r + 8*half
      const int bb = tok >> 11, nn = tok & (N_ - 1);
      const float v = acc[sub][r] + bv;
      if (t == 0)
        qws[((size_t)(bb * H_ + hh) * N_ + nn) * DH_ + d] = f2bf(v * 0.125f);
      else if (t == 1)
        kws[((size_t)(bb * H_ + hh) * N_ + nn) * DH_ + d] = f2bf(v);
      else
        vtws[((size_t)(bb * H_ + hh) * DH_ + d) * N_ + nn] = f2bf(v);
    }
  }
}

// ---------------------------------------------------------------------------
// Kernel 2: fused talking-heads attention. One WG = 16 query rows, all 8
// heads (wave g <-> head g). Two streaming passes over K/V (everything is
// L2-resident: K+V bf16 = 16MB << 192MB L2), cross-head mixing via LDS.
// ---------------------------------------------------------------------------
__global__ __launch_bounds__(256) void attn_kernel(
    const unsigned short* __restrict__ qw,
    const unsigned short* __restrict__ kw,
    const unsigned short* __restrict__ vtw,
    const float* __restrict__ th1g, const float* __restrict__ th2g,
    unsigned short* __restrict__ ohw) {
  __shared__ float          Sx[H_][16][68];   // raw per-head logits (padded)
  __shared__ unsigned short Px[H_][16][72];   // normalized probs, bf16 (padded)

  const int b    = blockIdx.x >> 7;
  const int q0   = (blockIdx.x & 127) << 4;
  const int wave = threadIdx.x >> 5;          // head id
  const int lane = threadIdx.x & 31;
  const int hsel = lane >> 4;
  const int row  = lane & 15;
  const int rl   = lane >> 1;                 // softmax: lane pair owns a row
  const int ch   = lane & 1;                  // column half (32 cols each)

  float th1r[8], th2r[8];
#pragma unroll
  for (int h = 0; h < 8; ++h) {
    th1r[h] = th1g[wave * 8 + h];
    th2r[h] = th2g[wave * 8 + h];
  }

  // Resident Q fragments for this head (K-dim = d, 2 steps of 32)
  Op16 qa[2];
  {
    const unsigned short* qp =
        qw + ((size_t)(b * H_ + wave) * N_ + q0 + row) * DH_;
#pragma unroll
    for (int ds2 = 0; ds2 < 2; ++ds2) qa[ds2].v = load_frag(qp + ds2 * 32, hsel);
  }
  const unsigned short* kb0 = kw  + (size_t)(b * H_ + wave) * N_ * DH_;
  const unsigned short* vb0 = vtw + (size_t)(b * H_ + wave) * DH_ * N_;

  // S tile: wave computes its head's 16x64 logits into LDS
  auto compute_S = [&](int kb) {
    if (kb + 64 < N_)
      __builtin_prefetch(kb0 + (size_t)(kb + 64) * DH_ + lane * 32, 0, 0);
#pragma unroll
    for (int sub = 0; sub < 4; ++sub) {
      v8f acc = {};
      const unsigned short* kp = kb0 + (size_t)(kb + sub * 16 + row) * DH_;
#pragma unroll
      for (int ds2 = 0; ds2 < 2; ++ds2) {
        v16bf bo = load_frag(kp + ds2 * 32, hsel);
        acc = __builtin_amdgcn_wmma_f32_16x16x32_bf16(
            false, qa[ds2].v, false, bo, (short)0, acc, false, false);
      }
#pragma unroll
      for (int r = 0; r < 8; ++r)
        Sx[wave][r + hsel * 8][sub * 16 + row] = acc[r];
    }
  };

  // th1-mix all heads into this wave's 32 owned elements (row rl, half ch)
  auto mix_th1 = [&](float* sp) {
#pragma unroll
    for (int i = 0; i < 32; ++i) sp[i] = 0.0f;
#pragma unroll
    for (int h = 0; h < 8; ++h) {
      const float w = th1r[h];
      const float4* src = (const float4*)&Sx[h][rl][ch * 32];
#pragma unroll
      for (int i = 0; i < 8; ++i) {
        float4 v4 = src[i];
        sp[4 * i + 0] += w * v4.x; sp[4 * i + 1] += w * v4.y;
        sp[4 * i + 2] += w * v4.z; sp[4 * i + 3] += w * v4.w;
      }
    }
  };

  float m_run = -3.0e38f, l_run = 0.0f;

  // -------- pass 1: row max & sum of exp of mixed logits --------
  for (int kb = 0; kb < N_; kb += 64) {
    compute_S(kb);
    __syncthreads();
    float sp[32];
    mix_th1(sp);
    float tmax = sp[0];
#pragma unroll
    for (int i = 1; i < 32; ++i) tmax = fmaxf(tmax, sp[i]);
    tmax = fmaxf(tmax, __shfl_xor(tmax, 1, 32));
    const float mnew = fmaxf(m_run, tmax);
    float psum = 0.0f;
#pragma unroll
    for (int i = 0; i < 32; ++i) psum += __expf(sp[i] - mnew);
    psum += __shfl_xor(psum, 1, 32);
    l_run = l_run * __expf(m_run - mnew) + psum;
    m_run = mnew;
    __syncthreads();
  }
  const float rcpl = 1.0f / l_run;

  v8f oacc[4] = {};   // 16 x 64 output accumulator for head `wave`

  // -------- pass 2: P = softmax, th2-mix, multiply by V --------
  for (int kb = 0; kb < N_; kb += 64) {
    compute_S(kb);
    __syncthreads();
    float sp[32];
    mix_th1(sp);
#pragma unroll
    for (int j = 0; j < 4; ++j) {
      u16x8 pk;
#pragma unroll
      for (int e = 0; e < 8; ++e)
        pk[e] = f2bf(__expf(sp[j * 8 + e] - m_run) * rcpl);
      *(u16x8*)&Px[wave][rl][ch * 32 + j * 8] = pk;
    }
    __syncthreads();
#pragma unroll
    for (int ks = 0; ks < 2; ++ks) {   // key chunk of 32 (WMMA K-dim)
      float mixv[16];
#pragma unroll
      for (int i = 0; i < 16; ++i) mixv[i] = 0.0f;
#pragma unroll
      for (int h = 0; h < 8; ++h) {    // th2 mix in A-fragment layout
        const float w = th2r[h];
        const unsigned short* pp = &Px[h][row][ks * 32 + hsel * 8];
        u16x8 a0 = *(const u16x8*)pp;
        u16x8 a1 = *(const u16x8*)(pp + 16);
#pragma unroll
        for (int i = 0; i < 8; ++i) mixv[i]     += w * bf2f(a0[i]);
#pragma unroll
        for (int i = 0; i < 8; ++i) mixv[8 + i] += w * bf2f(a1[i]);
      }
      Op16 ao;
#pragma unroll
      for (int i = 0; i < 16; ++i) ao.s[i] = f2bf(mixv[i]);
#pragma unroll
      for (int sub = 0; sub < 4; ++sub) {  // d-subtiles, V^T is key-contiguous
        v16bf bo = load_frag(
            vb0 + (size_t)(sub * 16 + row) * N_ + kb + ks * 32, hsel);
        oacc[sub] = __builtin_amdgcn_wmma_f32_16x16x32_bf16(
            false, ao.v, false, bo, (short)0, oacc[sub], false, false);
      }
    }
    __syncthreads();
  }

  // epilogue: bf16 head-output, token-major [b*n][h*64+d]
#pragma unroll
  for (int sub = 0; sub < 4; ++sub)
#pragma unroll
    for (int r = 0; r < 8; ++r) {
      const int m = q0 + r + hsel * 8;
      const int c = wave * 64 + sub * 16 + row;
      ohw[(size_t)(b * N_ + m) * DIM_ + c] = f2bf(oacc[sub][r]);
    }
}

// ---------------------------------------------------------------------------
// Kernel 3: output projection [8192x512]x[512x512] + bias -> f32 d_out.
// ---------------------------------------------------------------------------
__global__ __launch_bounds__(256) void out_gemm(
    const unsigned short* __restrict__ ohbf,
    const unsigned short* __restrict__ wT,
    const float* __restrict__ bias, float* __restrict__ out) {
  const int wave = threadIdx.x >> 5;
  const int lane = threadIdx.x & 31;
  const int hsel = lane >> 4;
  const int row  = lane & 15;
  const int tid  = blockIdx.x * 8 + wave;
  const int m0   = (tid / 8) * 16;
  const int n0   = (tid % 8) * 64;

  v8f acc[4] = {};
  const unsigned short* arow = ohbf + (size_t)(m0 + row) * DIM_;
  for (int k0 = 0; k0 < DIM_; k0 += 32) {
    v16bf a = load_frag(arow + k0, hsel);
#pragma unroll
    for (int sub = 0; sub < 4; ++sub) {
      v16bf b = load_frag(wT + (size_t)(n0 + sub * 16 + row) * DIM_ + k0, hsel);
      acc[sub] = __builtin_amdgcn_wmma_f32_16x16x32_bf16(
          false, a, false, b, (short)0, acc[sub], false, false);
    }
  }
#pragma unroll
  for (int sub = 0; sub < 4; ++sub) {
    const int c = n0 + sub * 16 + row;
    const float bv = bias[c];
#pragma unroll
    for (int r = 0; r < 8; ++r)
      out[(size_t)(m0 + r + hsel * 8) * DIM_ + c] = acc[sub][r] + bv;
  }
}

// ---------------------------------------------------------------------------
extern "C" void kernel_launch(void* const* d_in, const int* in_sizes, int n_in,
                              void* d_out, int out_size, void* d_ws,
                              size_t ws_size, hipStream_t stream) {
  (void)in_sizes; (void)n_in; (void)out_size; (void)ws_size;
  const float* x    = (const float*)d_in[0];
  const float* wqkv = (const float*)d_in[1];
  const float* bqkv = (const float*)d_in[2];
  const float* th1  = (const float*)d_in[3];
  const float* th2  = (const float*)d_in[4];
  const float* wout = (const float*)d_in[5];
  const float* bout = (const float*)d_in[6];
  float* out = (float*)d_out;

  char* ws = (char*)d_ws;                              // byte offsets:
  unsigned short* xbf   = (unsigned short*)(ws + 0);          //  8.0 MB
  unsigned short* wqkvT = (unsigned short*)(ws + 8388608);    //  1.5 MB
  unsigned short* woutT = (unsigned short*)(ws + 9961472);    //  0.5 MB
  unsigned short* qws   = (unsigned short*)(ws + 10485760);   //  8.0 MB
  unsigned short* kws   = (unsigned short*)(ws + 18874368);   //  8.0 MB
  unsigned short* vtws  = (unsigned short*)(ws + 27262976);   //  8.0 MB
  unsigned short* ohws  = (unsigned short*)(ws + 35651584);   //  8.0 MB (end 42MB)

  prep_kernel<<<2048, 256, 0, stream>>>(x, wqkv, wout, xbf, wqkvT, woutT);
  qkv_gemm<<<1536, 256, 0, stream>>>(xbf, wqkvT, bqkv, qws, kws, vtws);
  attn_kernel<<<512, 256, 0, stream>>>(qws, kws, vtws, th1, th2, ohws);
  out_gemm<<<512, 256, 0, stream>>>(ohws, woutT, bout, out);
}
